// _AttentionBlock_22797686407905
// MI455X (gfx1250) — matrix operand order repaired and checked by reading, not compiled, not run on hardware
//
#include <hip/hip_runtime.h>

#define BATCH 2
#define CH    512
#define SEQ   4096
#define NGROUP 32
#define EPSV  1e-5f

typedef __attribute__((ext_vector_type(16))) __bf16 bf16x16;
typedef __attribute__((ext_vector_type(8)))  __bf16 bf16x8;
typedef __attribute__((ext_vector_type(8)))  float  f32x8;
typedef unsigned short u16;

// ---- optional CDNA5 async memory->LDS path (ASYNCcnt tracked) --------------
#if defined(__gfx1250__) && __has_builtin(__builtin_amdgcn_global_load_async_to_lds_b128)
#define HAVE_ASYNC 1
typedef int v4i_t __attribute__((vector_size(4 * sizeof(int))));
typedef __attribute__((address_space(1))) v4i_t* g4ptr_t;   // global src
typedef __attribute__((address_space(3))) v4i_t* l4ptr_t;   // LDS dst
__device__ __forceinline__ void async_cp16(void* lds, const void* g) {
  __builtin_amdgcn_global_load_async_to_lds_b128(
      (g4ptr_t)(unsigned long long)(size_t)g,
      (l4ptr_t)(unsigned int)(size_t)lds, 0, 0);
}
__device__ __forceinline__ void async_wait0() {
#if __has_builtin(__builtin_amdgcn_s_wait_asynccnt)
  __builtin_amdgcn_s_wait_asynccnt(0);
#else
  asm volatile("s_wait_asynccnt 0x0" ::: "memory");
#endif
}
#else
#define HAVE_ASYNC 0
#endif

// float -> bf16, round-to-nearest-even
__device__ __forceinline__ u16 f2bf(float f) {
  unsigned int u = __float_as_uint(f);
  u += 0x7FFFu + ((u >> 16) & 1u);
  return (u16)(u >> 16);
}

// 16-element bf16 fragment from two 16-byte runs (CDNA5 16-bit A/B layouts:
// each lane holds two 8-element K runs; here all callers load one 16-run, k = 16*half + s).
__device__ __forceinline__ bf16x16 ld_frag(const u16* p0, const u16* p1) {
  bf16x8 lo = *(const bf16x8*)p0;
  bf16x8 hi = *(const bf16x8*)p1;
  return __builtin_shufflevector(lo, hi, 0,1,2,3,4,5,6,7,8,9,10,11,12,13,14,15);
}

__device__ __forceinline__ f32x8 wmma_bf16(bf16x16 a, bf16x16 b, f32x8 c) {
  return __builtin_amdgcn_wmma_f32_16x16x32_bf16(false, a, false, b, (short)0, c,
                                                 false, false);
}

// ---------------------------------------------------------------- weights cvt
__global__ void cvt_bf16(const float* __restrict__ src, u16* __restrict__ dst, int n) {
  int i = blockIdx.x * blockDim.x + threadIdx.x;
  int stride = gridDim.x * blockDim.x;
  for (; i < n; i += stride) dst[i] = f2bf(src[i]);
}

// ---------------------------------------------------------------- groupnorm
__global__ __launch_bounds__(256)
void gn_stats(const float* __restrict__ X, float* __restrict__ mean,
              float* __restrict__ rstd) {
  __shared__ float s1[256], s2[256];
  const int bg = blockIdx.x;
  const float* base = X + (size_t)bg * (16u * SEQ);  // group = 16 contig channels
  float sum = 0.f, sq = 0.f;
  for (int i = threadIdx.x; i < (16 * SEQ) / 4; i += 256) {
    float4 v = ((const float4*)base)[i];
    sum += v.x + v.y + v.z + v.w;
    sq  += v.x * v.x + v.y * v.y + v.z * v.z + v.w * v.w;
  }
  s1[threadIdx.x] = sum; s2[threadIdx.x] = sq;
  __syncthreads();
  for (int st = 128; st > 0; st >>= 1) {
    if (threadIdx.x < st) {
      s1[threadIdx.x] += s1[threadIdx.x + st];
      s2[threadIdx.x] += s2[threadIdx.x + st];
    }
    __syncthreads();
  }
  if (threadIdx.x == 0) {
    const float inv = 1.0f / (16.0f * SEQ);
    float mu  = s1[0] * inv;
    float var = s2[0] * inv - mu * mu;
    mean[bg] = mu;
    rstd[bg] = rsqrtf(var + EPSV);
  }
}

// Normalize + transpose [B,C,S] -> [B,S,C] bf16 via 64x64 LDS tile.
__global__ __launch_bounds__(256)
void gn_apply_t(const float* __restrict__ X, const float* __restrict__ gw,
                const float* __restrict__ gb, const float* __restrict__ mean,
                const float* __restrict__ rstd, u16* __restrict__ H) {
  __shared__ __align__(16) u16 tile[64 * 72];
  const int bb = blockIdx.z;
  const int s0 = blockIdx.y * 64;
  const int c0 = blockIdx.x * 64;
  const int tid = threadIdx.x;
  {
    int sx = tid & 63, cy = tid >> 6;
    for (int i = 0; i < 16; ++i) {
      int c  = cy + i * 4;
      int cg = c0 + c;
      int g  = bb * NGROUP + (cg >> 4);
      float mu = mean[g], rs = rstd[g];
      float v  = X[((size_t)bb * CH + cg) * SEQ + s0 + sx];
      float hn = (v - mu) * rs * gw[cg] + gb[cg];
      tile[c * 72 + sx] = f2bf(hn);
    }
  }
  __syncthreads();
  {
    int cx = tid & 63, sy = tid >> 6;
    for (int i = 0; i < 16; ++i) {
      int s = sy + i * 4;
      H[((size_t)bb * SEQ + s0 + s) * CH + c0 + cx] = tile[cx * 72 + s];
    }
  }
}

// ---------------------------------------------------------------- WMMA GEMM
// out = alpha * (A @ W^T + bias)   A:[B,M,K] bf16, W:[N,K] bf16
// MODE 0: bf16 [B,M,N]   MODE 1: bf16 [B,N,M]   MODE 2: f32 [B,N,M] + resid
template <int MODE>
__global__ __launch_bounds__(256)
void gemm_nt(const u16* __restrict__ A, const u16* __restrict__ Wt,
             const float* __restrict__ bias, u16* __restrict__ outB,
             float* __restrict__ outF, const float* __restrict__ resid,
             float alpha, int M, int N, int K) {
  const int bb = blockIdx.z;
  const int m0 = blockIdx.y * 64;
  const int n0 = blockIdx.x * 64;
  const int tid = threadIdx.x;
  const int w = tid >> 5, lane = tid & 31;
  const int lsub = lane & 15, half = lane >> 4;
  const int mBase = m0 + (w & 3) * 16;
  const int nBase = n0 + (w >> 2) * 32;

  f32x8 acc0 = {}, acc1 = {};
  const u16* Arow  = A  + ((size_t)bb * M + (mBase + lsub)) * K;
  const u16* Brow0 = Wt + (size_t)(nBase + lsub) * K;
  const u16* Brow1 = Wt + (size_t)(nBase + 16 + lsub) * K;
  for (int k0 = 0; k0 < K; k0 += 32) {
    bf16x16 af  = ld_frag(Arow  + k0 + 16 * half, Arow  + k0 + 16 * half + 8);
    bf16x16 bf0 = ld_frag(Brow0 + k0 + 16 * half, Brow0 + k0 + 16 * half + 8);
    bf16x16 bf1 = ld_frag(Brow1 + k0 + 16 * half, Brow1 + k0 + 16 * half + 8);
    acc0 = wmma_bf16(af, bf0, acc0);
    acc1 = wmma_bf16(af, bf1, acc1);
  }
  for (int t = 0; t < 2; ++t) {
    f32x8 acc = t ? acc1 : acc0;
    int n_abs = nBase + t * 16 + lsub;
    float bv = bias[n_abs];
    for (int i = 0; i < 8; ++i) {
      int m_abs = mBase + i + 8 * half;
      float val = alpha * (acc[i] + bv);
      if (MODE == 0) {
        outB[((size_t)bb * M + m_abs) * N + n_abs] = f2bf(val);
      } else if (MODE == 1) {
        outB[((size_t)bb * N + n_abs) * M + m_abs] = f2bf(val);
      } else {
        size_t ix = ((size_t)bb * N + n_abs) * M + m_abs;
        outF[ix] = val + resid[ix];
      }
    }
  }
}

// ---------------------------------------------------------------- flash attn
// Q,K: [B,S,C] bf16 (Q pre-scaled). Vt: [B,C,S] bf16. O: [B,S,C] bf16.
// Block: 64 query rows, 8 waves, dynamic LDS (~90KB of the 320KB WGP pool).
#define FLASH_LDS (64*512*2 + 64*65*4 + 64*72*2 + 3*64*4)  // 92160 B

__global__ __launch_bounds__(256)
void flash_attn(const u16* __restrict__ Q, const u16* __restrict__ Kk,
                const u16* __restrict__ Vt, u16* __restrict__ O) {
  extern __shared__ __align__(16) char smem[];
  u16*   q_s  = (u16*)smem;                       // 64 x 512
  float* s_s  = (float*)(smem + 65536);           // 64 x 65 (padded)
  u16*   p_s  = (u16*)(smem + 65536 + 16640);     // 64 x 72 (padded)
  float* m_sL = (float*)(smem + 65536 + 16640 + 9216);
  float* l_sL = m_sL + 64;
  float* a_sL = l_sL + 64;

  const int bb = blockIdx.y;
  const int q0 = blockIdx.x * 64;
  const int tid = threadIdx.x;
  const int w = tid >> 5, lane = tid & 31;
  const int lsub = lane & 15, half = lane >> 4;

  // ---- stage Q tile (64x512 bf16 = 64KB) into LDS
  {
    const u16* src = Q + ((size_t)bb * SEQ + q0) * CH;
#if HAVE_ASYNC
    for (int i = tid; i < (64 * 512) / 8; i += 256)
      async_cp16(q_s + i * 8, src + i * 8);
    async_wait0();
#else
    for (int i = tid; i < (64 * 512) / 8; i += 256)
      *(bf16x8*)(q_s + i * 8) = *(const bf16x8*)(src + i * 8);
#endif
    if (tid < 64) { m_sL[tid] = -3.0e38f; l_sL[tid] = 0.f; }
  }
  __syncthreads();

  const f32x8 vzero = {};
  f32x8 oacc[4][4];
#pragma unroll
  for (int a = 0; a < 4; ++a)
#pragma unroll
    for (int b2 = 0; b2 < 4; ++b2) oacc[a][b2] = vzero;

  const int mt = w & 3, np = w >> 2;
  for (int j = 0; j < SEQ / 64; ++j) {
    const int t0 = j * 64;
    // ---- S = Q @ K^T : two 16x16 tiles per wave
    f32x8 sa0 = {}, sa1 = {};
    const u16* qrow  = q_s + (mt * 16 + lsub) * 512;
    const u16* krow0 = Kk + ((size_t)bb * SEQ + (t0 + np * 32 + lsub)) * CH;
    const u16* krow1 = krow0 + (size_t)16 * CH;
    // prefetch next key/value blocks while this one computes
    __builtin_prefetch(krow0 + (size_t)64 * CH, 0, 1);
    __builtin_prefetch(Vt + ((size_t)bb * CH + (w * 64 + lsub)) * SEQ + t0 + 64, 0, 1);
#pragma unroll
    for (int k0 = 0; k0 < 512; k0 += 32) {
      bf16x16 af = ld_frag(qrow + k0 + 16 * half, qrow + k0 + 16 * half + 8);
      bf16x16 b0 = ld_frag(krow0 + k0 + 16 * half, krow0 + k0 + 16 * half + 8);
      bf16x16 b1 = ld_frag(krow1 + k0 + 16 * half, krow1 + k0 + 16 * half + 8);
      sa0 = wmma_bf16(af, b0, sa0);
      sa1 = wmma_bf16(af, b1, sa1);
    }
#pragma unroll
    for (int i = 0; i < 8; ++i) {
      int r = mt * 16 + i + 8 * half;
      s_s[r * 65 + np * 32 + lsub]      = sa0[i];
      s_s[r * 65 + np * 32 + 16 + lsub] = sa1[i];
    }
    __syncthreads();

    // ---- online softmax: 4 threads per row, wave32 lane permutes to combine
    {
      int r = tid >> 2, part = tid & 3;
      const float* srow = s_s + r * 65 + part * 16;
      float mx = -3.0e38f;
#pragma unroll
      for (int i = 0; i < 16; ++i) mx = fmaxf(mx, srow[i]);
      mx = fmaxf(mx, __shfl_xor(mx, 1));
      mx = fmaxf(mx, __shfl_xor(mx, 2));
      float mo = m_sL[r];
      mx = fmaxf(mx, mo);
      float sum = 0.f;
      u16* prow = p_s + r * 72 + part * 16;
#pragma unroll
      for (int i = 0; i < 16; ++i) {
        float p = __expf(srow[i] - mx);
        prow[i] = f2bf(p);
        sum += p;
      }
      sum += __shfl_xor(sum, 1);
      sum += __shfl_xor(sum, 2);
      if (part == 0) {
        float al = __expf(mo - mx);
        m_sL[r] = mx;
        l_sL[r] = l_sL[r] * al + sum;
        a_sL[r] = al;
      }
    }
    __syncthreads();

    // ---- rescale running O, then O += P @ V (wave's 64-wide c-slice)
#pragma unroll
    for (int tm = 0; tm < 4; ++tm)
#pragma unroll
      for (int i = 0; i < 8; ++i) {
        float al = a_sL[tm * 16 + i + 8 * half];
#pragma unroll
        for (int tn = 0; tn < 4; ++tn) oacc[tm][tn][i] *= al;
      }
#pragma unroll
    for (int kt = 0; kt < 2; ++kt) {
      bf16x16 vf[4];
#pragma unroll
      for (int tn = 0; tn < 4; ++tn) {
        const u16* vrow = Vt + ((size_t)bb * CH + (w * 64 + tn * 16 + lsub)) * SEQ
                             + t0 + kt * 32 + 16 * half;
        vf[tn] = ld_frag(vrow, vrow + 8);
      }
#pragma unroll
      for (int tm = 0; tm < 4; ++tm) {
        const u16* prow = p_s + (tm * 16 + lsub) * 72 + kt * 32;
        bf16x16 pf = ld_frag(prow + 16 * half, prow + 16 * half + 8);
#pragma unroll
        for (int tn = 0; tn < 4; ++tn)
          oacc[tm][tn] = wmma_bf16(pf, vf[tn], oacc[tm][tn]);
      }
    }
    __syncthreads();
  }

  // ---- finalize: O / l
#pragma unroll
  for (int tm = 0; tm < 4; ++tm)
#pragma unroll
    for (int i = 0; i < 8; ++i) {
      int r = tm * 16 + i + 8 * half;
      float linv = 1.0f / l_sL[r];
#pragma unroll
      for (int tn = 0; tn < 4; ++tn)
        O[((size_t)bb * SEQ + q0 + r) * CH + w * 64 + tn * 16 + lsub] =
            f2bf(oacc[tm][tn][i] * linv);
    }
}

// ---------------------------------------------------------------- launch
extern "C" void kernel_launch(void* const* d_in, const int* in_sizes, int n_in,
                              void* d_out, int out_size, void* d_ws, size_t ws_size,
                              hipStream_t stream) {
  const float* x  = (const float*)d_in[0];
  const float* gw = (const float*)d_in[1];
  const float* gb = (const float*)d_in[2];
  const float* wq = (const float*)d_in[3];
  const float* bq = (const float*)d_in[4];
  const float* wk = (const float*)d_in[5];
  const float* bk = (const float*)d_in[6];
  const float* wv = (const float*)d_in[7];
  const float* bv = (const float*)d_in[8];
  const float* wo = (const float*)d_in[9];
  const float* bo = (const float*)d_in[10];
  float* out = (float*)d_out;

  char* ws = (char*)d_ws;
  const size_t WMAT = (size_t)CH * CH;          // 262144 elems
  const size_t HMAT = (size_t)BATCH * SEQ * CH; // 4194304 elems
  u16* wq_b = (u16*)(ws);
  u16* wk_b = wq_b + WMAT;
  u16* wv_b = wk_b + WMAT;
  u16* wo_b = wv_b + WMAT;
  float* mean = (float*)(wo_b + WMAT);          // 64 floats
  float* rstd = mean + 64;                      // 64 floats
  u16* h_b  = (u16*)(ws + 4 * WMAT * sizeof(u16) + 1024);
  u16* q_b  = h_b + HMAT;
  u16* k_b  = q_b + HMAT;
  u16* vT_b = k_b + HMAT;
  u16* o_b  = vT_b + HMAT;

  // 1) weights -> bf16
  cvt_bf16<<<256, 256, 0, stream>>>(wq, wq_b, (int)WMAT);
  cvt_bf16<<<256, 256, 0, stream>>>(wk, wk_b, (int)WMAT);
  cvt_bf16<<<256, 256, 0, stream>>>(wv, wv_b, (int)WMAT);
  cvt_bf16<<<256, 256, 0, stream>>>(wo, wo_b, (int)WMAT);

  // 2) groupnorm stats + apply/transpose -> h_b [B,S,C] bf16
  gn_stats<<<BATCH * NGROUP, 256, 0, stream>>>(x, mean, rstd);
  gn_apply_t<<<dim3(CH / 64, SEQ / 64, BATCH), 256, 0, stream>>>(
      x, gw, gb, mean, rstd, h_b);

  // 3) QKV projections (scale folded into Q; V stored transposed)
  const float scale = 0.04419417382415922f;  // 512^-0.5
  dim3 ggrid(CH / 64, SEQ / 64, BATCH);
  gemm_nt<0><<<ggrid, 256, 0, stream>>>(h_b, wq_b, bq, q_b, nullptr, nullptr,
                                        scale, SEQ, CH, CH);
  gemm_nt<0><<<ggrid, 256, 0, stream>>>(h_b, wk_b, bk, k_b, nullptr, nullptr,
                                        1.0f, SEQ, CH, CH);
  gemm_nt<1><<<ggrid, 256, 0, stream>>>(h_b, wv_b, bv, vT_b, nullptr, nullptr,
                                        1.0f, SEQ, CH, CH);

  // 4) flash attention (Bq=64) -> o_b [B,S,C] bf16
  flash_attn<<<dim3(SEQ / 64, BATCH), 256, FLASH_LDS, stream>>>(q_b, k_b, vT_b, o_b);

  // 5) output projection + bias + residual, store [B,C,S] f32
  gemm_nt<2><<<ggrid, 256, 0, stream>>>(o_b, wo_b, bo, nullptr, out, x,
                                        1.0f, SEQ, CH, CH);
}
